// MPWEncoder_55619826483351
// MI455X (gfx1250) — compile-verified, matrix operand order checked by reading, not run on hardware
//
#include <hip/hip_runtime.h>
#include <hip/hip_bf16.h>
#include <math.h>

// ---------------------------------------------------------------------------
// MPWEncoder: tanh( tanh( tanh(x @ (Wih*Mih)^T + bih) @ (Whc*Mhc)^T + bhc )
//                   |> conv1d(k2,s2) |> tanh |> conv1d(k2,s2) ) on MI455X.
// bf16 WMMA (v_wmma_f32_16x16x32_bf16) for both GEMMs, hw v_tanh_f32 where
// available, vectorized b128 packing prolog, fused double-conv epilogue.
// ---------------------------------------------------------------------------

#define IN_DIM   4096
#define HID_DIM  2048
#define CNV1_DIM 1024
#define BATCH    16384

typedef __bf16 bf16x16 __attribute__((ext_vector_type(16)));
typedef float  f32x8   __attribute__((ext_vector_type(8)));

// gfx1250 has a hardware transcendental TANH (TRANS32 class, co-executes
// with WMMA). Use the builtin if this clang declares it; else fall back.
#if __has_builtin(__builtin_amdgcn_tanhf)
static __device__ __forceinline__ float fast_tanh(float x) { return __builtin_amdgcn_tanhf(x); }
#elif __has_builtin(__builtin_amdgcn_tanh_f32)
static __device__ __forceinline__ float fast_tanh(float x) { return __builtin_amdgcn_tanh_f32(x); }
#else
static __device__ __forceinline__ float fast_tanh(float x) { return tanhf(x); }
#endif

static __device__ __forceinline__ unsigned short f32_to_bf16_rn(float f) {
    unsigned int u = __float_as_uint(f);
    u += 0x7FFFu + ((u >> 16) & 1u);          // round-to-nearest-even
    return (unsigned short)(u >> 16);
}

// ------------------------------- packing -----------------------------------
// 8 elements per thread-iteration: 2x float4 (b128) loads -> 1x uint4 store.

__global__ __launch_bounds__(256)
void pack_bf16_kernel(const float* __restrict__ s, unsigned short* __restrict__ d, size_t n8) {
    size_t i = (size_t)blockIdx.x * blockDim.x + threadIdx.x;
    size_t stride = (size_t)gridDim.x * blockDim.x;
    for (; i < n8; i += stride) {
        const float4 a = *(const float4*)(s + i * 8);
        const float4 b = *(const float4*)(s + i * 8 + 4);
        union { unsigned short h[8]; uint4 u; } o;
        o.h[0] = f32_to_bf16_rn(a.x); o.h[1] = f32_to_bf16_rn(a.y);
        o.h[2] = f32_to_bf16_rn(a.z); o.h[3] = f32_to_bf16_rn(a.w);
        o.h[4] = f32_to_bf16_rn(b.x); o.h[5] = f32_to_bf16_rn(b.y);
        o.h[6] = f32_to_bf16_rn(b.z); o.h[7] = f32_to_bf16_rn(b.w);
        *(uint4*)(d + i * 8) = o.u;
    }
}

__global__ __launch_bounds__(256)
void pack_masked_bf16_kernel(const float* __restrict__ w, const unsigned char* __restrict__ m,
                             unsigned short* __restrict__ d, size_t n8) {
    size_t i = (size_t)blockIdx.x * blockDim.x + threadIdx.x;
    size_t stride = (size_t)gridDim.x * blockDim.x;
    for (; i < n8; i += stride) {
        const float4 a = *(const float4*)(w + i * 8);
        const float4 b = *(const float4*)(w + i * 8 + 4);
        union { unsigned char c[8]; uint2 u; } mk;
        mk.u = *(const uint2*)(m + i * 8);
        union { unsigned short h[8]; uint4 u; } o;
        o.h[0] = f32_to_bf16_rn(mk.c[0] ? a.x : 0.f);
        o.h[1] = f32_to_bf16_rn(mk.c[1] ? a.y : 0.f);
        o.h[2] = f32_to_bf16_rn(mk.c[2] ? a.z : 0.f);
        o.h[3] = f32_to_bf16_rn(mk.c[3] ? a.w : 0.f);
        o.h[4] = f32_to_bf16_rn(mk.c[4] ? b.x : 0.f);
        o.h[5] = f32_to_bf16_rn(mk.c[5] ? b.y : 0.f);
        o.h[6] = f32_to_bf16_rn(mk.c[6] ? b.z : 0.f);
        o.h[7] = f32_to_bf16_rn(mk.c[7] ? b.w : 0.f);
        *(uint4*)(d + i * 8) = o.u;
    }
}

// ------------------------------ WMMA GEMM ----------------------------------
// C[M,N] = tanh(A[M,K](bf16) * W[N,K](bf16)^T + bias[N]); OUT_BF16 picks the
// output element type. Tiles: BM=BN=128, BK=32, 256 threads = 8 waves (4x2).

#define BM 128
#define BN 128
#define BK 32
#define LDST 40   // LDS row stride in shorts: 80B => 20-bank stride, 16B aligned

template <bool OUT_BF16>
__global__ __launch_bounds__(256)
void wmma_gemm_bias_tanh(const unsigned short* __restrict__ A,   // [M,K] bf16 bits
                         const unsigned short* __restrict__ W,   // [N,K] bf16 bits
                         const float* __restrict__ bias,         // [N]
                         void* __restrict__ out, int M, int N, int K) {
    __shared__ __align__(16) unsigned short lA[BM * LDST];
    __shared__ __align__(16) unsigned short lB[BN * LDST];

    const int tid   = threadIdx.x;
    const int lane  = tid & 31;
    const int wave  = tid >> 5;
    const int waveM = wave >> 1;      // 0..3 -> 32 rows each
    const int waveN = wave & 1;       // 0..1 -> 64 cols each
    const int l15   = lane & 15;
    const int ksel  = lane >> 4;      // ISA 16-bit A-frag half-K selector
    const int blockM0 = blockIdx.y * BM;
    const int blockN0 = blockIdx.x * BN;

    f32x8 acc[2][4];
#pragma unroll
    for (int mi = 0; mi < 2; ++mi)
#pragma unroll
        for (int ni = 0; ni < 4; ++ni)
            acc[mi][ni] = (f32x8){0.f, 0.f, 0.f, 0.f, 0.f, 0.f, 0.f, 0.f};

    for (int k0 = 0; k0 < K; k0 += BK) {
        // stage global -> regs (2 x 16B chunks per thread per matrix)
        uint4 ra[2], rb[2];
#pragma unroll
        for (int i = 0; i < 2; ++i) {
            int c = tid + i * 256;
            int row = c >> 2, sub = c & 3;              // 128 rows x 4 chunks
            ra[i] = *(const uint4*)(A + (size_t)(blockM0 + row) * K + k0 + sub * 8);
            rb[i] = *(const uint4*)(W + (size_t)(blockN0 + row) * K + k0 + sub * 8);
        }
        __syncthreads();                                // prev-iter LDS reads done
#pragma unroll
        for (int i = 0; i < 2; ++i) {
            int c = tid + i * 256;
            int row = c >> 2, sub = c & 3;
            *(uint4*)(&lA[row * LDST + sub * 8]) = ra[i];
            *(uint4*)(&lB[row * LDST + sub * 8]) = rb[i];
        }
        __syncthreads();

        // fragment loads per ISA 16-bit layout:
        // lane<16: K {0..7, 16..23}; lane>=16: K {8..15, 24..31}
        union FragU { bf16x16 v; uint4 u[2]; } aF[2], bF[4];
#pragma unroll
        for (int mi = 0; mi < 2; ++mi) {
            int r = waveM * 32 + mi * 16 + l15;
            aF[mi].u[0] = *(const uint4*)(&lA[r * LDST + ksel * 8]);
            aF[mi].u[1] = *(const uint4*)(&lA[r * LDST + 16 + ksel * 8]);
        }
#pragma unroll
        for (int ni = 0; ni < 4; ++ni) {
            int r = waveN * 64 + ni * 16 + l15;
            bF[ni].u[0] = *(const uint4*)(&lB[r * LDST + ksel * 8]);
            bF[ni].u[1] = *(const uint4*)(&lB[r * LDST + 16 + ksel * 8]);
        }
#pragma unroll
        for (int mi = 0; mi < 2; ++mi)
#pragma unroll
            for (int ni = 0; ni < 4; ++ni)
                acc[mi][ni] = __builtin_amdgcn_wmma_f32_16x16x32_bf16(
                    false, aF[mi].v, false, bF[ni].v,
                    (short)0, acc[mi][ni], false, false);
    }

    // epilogue: D layout VGPR r -> row r + 8*ksel, col = lane&15
#pragma unroll
    for (int mi = 0; mi < 2; ++mi) {
        int mbase = blockM0 + waveM * 32 + mi * 16 + 8 * ksel;
#pragma unroll
        for (int ni = 0; ni < 4; ++ni) {
            int col = blockN0 + waveN * 64 + ni * 16 + l15;
            float bv = bias[col];
#pragma unroll
            for (int r = 0; r < 8; ++r) {
                float v = fast_tanh(acc[mi][ni][r] + bv);
                size_t o = (size_t)(mbase + r) * N + col;
                if (OUT_BF16) ((unsigned short*)out)[o] = f32_to_bf16_rn(v);
                else          ((float*)out)[o] = v;
            }
        }
    }
}

// ----------------------- fused conv1d(k2,s2) x2 + tanh ---------------------
// y2[b,i] depends on c[b, 4i..4i+3] only.

__global__ __launch_bounds__(256)
void conv_fused_kernel(const float* __restrict__ c, float* __restrict__ y,
                       const float* __restrict__ wc1, const float* __restrict__ bc1,
                       const float* __restrict__ wc2, const float* __restrict__ bc2,
                       int total) {
    int idx = blockIdx.x * blockDim.x + threadIdx.x;
    if (idx >= total) return;
    int b = idx >> 8;          // CNV1_DIM/4 = 256 outputs per batch row
    int i = idx & 255;
    const float4 v = *(const float4*)(c + (size_t)b * CNV1_DIM + i * 4);
    float w10 = wc1[0], w11 = wc1[1], b1 = bc1[0];
    float w20 = wc2[0], w21 = wc2[1], b2 = bc2[0];
    float y1a = fast_tanh(w10 * v.x + w11 * v.y + b1);
    float y1b = fast_tanh(w10 * v.z + w11 * v.w + b1);
    y[idx] = fast_tanh(w20 * y1a + w21 * y1b + b2);
}

// --------------------------------- launch ----------------------------------

extern "C" void kernel_launch(void* const* d_in, const int* in_sizes, int n_in,
                              void* d_out, int out_size, void* d_ws, size_t ws_size,
                              hipStream_t stream) {
    const float*         x    = (const float*)d_in[0];
    const unsigned char* mih  = (const unsigned char*)d_in[1];   // bool mask
    const unsigned char* mhc  = (const unsigned char*)d_in[2];   // bool mask
    const float*         wih  = (const float*)d_in[3];
    const float*         bih  = (const float*)d_in[4];
    const float*         whc  = (const float*)d_in[5];
    const float*         bhc  = (const float*)d_in[6];
    const float*         wc1  = (const float*)d_in[7];
    const float*         bc1  = (const float*)d_in[8];
    const float*         wc2  = (const float*)d_in[9];
    const float*         bc2  = (const float*)d_in[10];
    float*               out  = (float*)d_out;

    const size_t nX   = (size_t)BATCH * IN_DIM;      // 67.1M
    const size_t nH   = (size_t)BATCH * HID_DIM;     // 33.6M
    const size_t nC   = (size_t)BATCH * CNV1_DIM;    // 16.8M
    const size_t nWih = (size_t)HID_DIM * IN_DIM;    // 8.39M
    const size_t nWhc = (size_t)CNV1_DIM * HID_DIM;  // 2.10M

    char* ws = (char*)d_ws;
    unsigned short* xb   = (unsigned short*)ws;                 ws += nX   * 2;
    unsigned short* hb   = (unsigned short*)ws;                 ws += nH   * 2;
    unsigned short* wihb = (unsigned short*)ws;                 ws += nWih * 2;
    unsigned short* whcb = (unsigned short*)ws;                 ws += nWhc * 2;
    float*          cf   = (float*)ws;                          ws += nC   * 4;

    // 1) pack inputs/weights to bf16 (memory-bound prolog, b128 vectorized)
    pack_bf16_kernel<<<4096, 256, 0, stream>>>(x, xb, nX / 8);
    pack_masked_bf16_kernel<<<2048, 256, 0, stream>>>(wih, mih, wihb, nWih / 8);
    pack_masked_bf16_kernel<<<1024, 256, 0, stream>>>(whc, mhc, whcb, nWhc / 8);

    // 2) h = tanh(x @ (Wih*M)^T + bih)   [BATCH, HID_DIM] bf16 out
    {
        dim3 grid(HID_DIM / BN, BATCH / BM);   // (16, 128)
        wmma_gemm_bias_tanh<true><<<grid, 256, 0, stream>>>(
            xb, wihb, bih, (void*)hb, BATCH, HID_DIM, IN_DIM);
    }

    // 3) c = tanh(h @ (Whc*M)^T + bhc)   [BATCH, CNV1_DIM] f32 out
    {
        dim3 grid(CNV1_DIM / BN, BATCH / BM);  // (8, 128)
        wmma_gemm_bias_tanh<false><<<grid, 256, 0, stream>>>(
            hb, whcb, bhc, (void*)cf, BATCH, CNV1_DIM, HID_DIM);
    }

    // 4) y = tanh(conv2(tanh(conv1(c))))  [BATCH, CNV1_DIM/4]
    {
        int total = BATCH * (CNV1_DIM / 4);    // == out_size
        conv_fused_kernel<<<(total + 255) / 256, 256, 0, stream>>>(
            cf, out, wc1, bc1, wc2, bc2, total);
    }
}
